// EMAQuantizer_35416300323332
// MI455X (gfx1250) — matrix operand order repaired and checked by reading, not compiled, not run on hardware
//
#include <hip/hip_runtime.h>
#include <hip/hip_bf16.h>

// Problem constants (fixed by the reference): B=32, C=512, H=W=32, K=1024
#define CDIM 512
#define KNUM 1024
#define NPIX 32768            // B*H*W
#define QELEMS 16777216       // B*C*H*W

// ---- toolchain feature gates -----------------------------------------------
#if defined(__has_builtin)
#if __has_builtin(__builtin_amdgcn_tensor_load_to_lds)
#define HAVE_TDM 1
#endif
#endif
#ifndef HAVE_TDM
#define HAVE_TDM 0
#endif
#if __has_include(<hip/amd_detail/amd_gfx1250_TDM.h>)
#define TDM_6ARG 1   // amdgpu-toolchain (clang-23 / therock): 6-arg builtin
#else
#define TDM_6ARG 0   // ROCm 7.2 (clang-22): 5-arg builtin
#endif

typedef __attribute__((ext_vector_type(16))) __bf16 v16bf;
typedef __attribute__((ext_vector_type(8)))  float  v8f;
typedef __attribute__((ext_vector_type(4)))  unsigned int v4u;
typedef __attribute__((ext_vector_type(8)))  int  v8i;
typedef __attribute__((ext_vector_type(4)))  int  v4i;

union FragBF {
    uint4 u[2];   // two 16-byte chunks -> 16 bf16 -> 8 VGPRs
    v16bf v;
};

// LDS geometry: rows of 512 bf16 (1024B) padded by +16B -> 4-bank shift per row
#define ROWB    1040u
#define B_BYTES (64u * ROWB)          // 66560: one 64-k embedding chunk
#define LDS_TOTAL (2u * B_BYTES)      // 133120 -> 2 blocks/WGP (<=320KB)

extern __shared__ char lds_raw[];

// ---------------------------------------------------------------------------
// TDM: DMA a [tile_rows x 512] bf16 tile (row-major, row stride 512) into LDS
// at byte offset lds_off, inserting 16B of LDS padding after every 1024B row.
// D# per CDNA5 ISA ch.8 (groups 0/1; groups 2/3 zero => 2D tensor).
// ---------------------------------------------------------------------------
#if HAVE_TDM
__device__ __forceinline__ void tdm_load_tile(unsigned lds_off, const void* gsrc,
                                              int tile_rows, int tensor_rows) {
    unsigned long long ga = (unsigned long long)gsrc;
    v4u g0;
    g0.x = 1u;                                            // count=1 (valid), user D#
    g0.y = lds_off;                                       // LDS byte address
    g0.z = (unsigned)ga;                                  // global_addr[31:0]
    g0.w = (unsigned)((ga >> 32) & 0x1FFFFFFu) | (2u << 30); // addr[56:32] | type=2
    v8i g1;
    g1[0] = (1 << 16)      // data_size = 2 bytes
          | (1 << 20)      // pad_enable
          | (7 << 22)      // pad_interval: every 256 dwords (=1024B row)
          | (3 << 25);     // pad_amount: 4 dwords (=16B)  -> 1040B LDS row stride
    g1[1] = (int)(512u << 16);                 // tensor_dim0 = 512 (lo16 @ [63:48])
    g1[2] = (int)(((unsigned)tensor_rows & 0xffffu) << 16); // dim0 hi=0 | tensor_dim1 lo
    g1[3] = (int)(512u << 16);                 // tensor_dim1 hi=0 | tile_dim0 = 512
    g1[4] = tile_rows;                         // tile_dim1 | tile_dim2=0
    g1[5] = 512;                               // tensor_dim0_stride[31:0] = 512
    g1[6] = 0;
    g1[7] = 0;
    v4i z4 = {0, 0, 0, 0};
#if TDM_6ARG
    v8i z8 = {};
    __builtin_amdgcn_tensor_load_to_lds(g0, g1, z4, z4, z8, 0);
#else
    __builtin_amdgcn_tensor_load_to_lds(g0, g1, z4, z4, 0);
#endif
}
#else
// Fallback: cooperative global->LDS staging with the same padded layout.
__device__ __forceinline__ void stage_tile(unsigned lds_off, const __bf16* g,
                                           int rows, int tid) {
    for (int r = tid >> 4; r < rows; r += 16) {          // 16 threads per 1024B row
        const uint4* src = (const uint4*)(g + (size_t)r * CDIM) + (tid & 15);
        uint4* dst = (uint4*)(lds_raw + lds_off + r * ROWB) + (tid & 15);
        #pragma unroll
        for (int j = 0; j < 4; j++) dst[j * 16] = src[j * 16];
    }
}
#endif

// ---------------------------------------------------------------------------
// Kernel 1: embedding -> bf16 row-major, plus ||e_k||^2
// ---------------------------------------------------------------------------
__global__ __launch_bounds__(128) void vq_prep_embed(const float* __restrict__ E,
                                                     __bf16* __restrict__ Eb,
                                                     float* __restrict__ enorm) {
    int k = blockIdx.x;
    const float* row = E + (size_t)k * CDIM;
    float s = 0.f;
    for (int c = threadIdx.x; c < CDIM; c += 128) {
        float v = row[c];
        Eb[(size_t)k * CDIM + c] = (__bf16)v;
        s += v * v;
    }
    #pragma unroll
    for (int off = 16; off; off >>= 1) s += __shfl_xor(s, off, 32);
    __shared__ float wsum[4];
    if ((threadIdx.x & 31) == 0) wsum[threadIdx.x >> 5] = s;
    __syncthreads();
    if (threadIdx.x == 0)
        enorm[k] = wsum[0] + wsum[1] + wsum[2] + wsum[3];
}

// ---------------------------------------------------------------------------
// Kernel 2: z NCHW f32 -> Zb[n][c] bf16 row-major via LDS 32x32 tile transpose
// ---------------------------------------------------------------------------
__global__ __launch_bounds__(256) void vq_zconv(const float* __restrict__ z,
                                                __bf16* __restrict__ Zb) {
    int b  = blockIdx.x;
    int c0 = blockIdx.y * 32;
    int p0 = blockIdx.z * 32;
    __shared__ float t[32][33];
    const float* zp = z + (size_t)b * CDIM * 1024;
    int tx = threadIdx.x & 31, ty = threadIdx.x >> 5;
    #pragma unroll
    for (int s = 0; s < 4; s++) {
        int ci = ty + s * 8;
        t[ci][tx] = zp[(size_t)(c0 + ci) * 1024 + p0 + tx];
    }
    __syncthreads();
    #pragma unroll
    for (int s = 0; s < 4; s++) {
        int pi = ty + s * 8;
        Zb[((size_t)b * 1024 + p0 + pi) * CDIM + c0 + tx] = (__bf16)t[tx][pi];
    }
}

// ---------------------------------------------------------------------------
// Kernel 3: GEMM + fused argmin.
// A panel (16 rows x 512 c per wave) lives entirely in VGPRs (128 regs),
// loaded once. B (embedding) chunks of 64 k double-buffered in LDS via TDM.
// Inner step: 8 batched ds_load_b128 -> 4 back-to-back WMMAs.
// ---------------------------------------------------------------------------
__global__ __launch_bounds__(256) void vq_main(const __bf16* __restrict__ Zb,
                                               const __bf16* __restrict__ Eb,
                                               const float* __restrict__ enorm,
                                               int* __restrict__ idx_out) {
    const int lane = threadIdx.x & 31;
    const int wave = threadIdx.x >> 5;
    const int m0   = blockIdx.x * 128 + wave * 16;
    const int half = lane >> 4;
    const int l16  = lane & 15;

    // Kick off first B chunk before touching A (overlap DMA with A loads).
#if HAVE_TDM
    if (wave == 0) tdm_load_tile(0u, Eb, 64, KNUM);
#else
    stage_tile(0u, Eb, 64, threadIdx.x);
#endif

    // --- A panel in registers: ISA 16-bit A 16x32 layout per c-step --------
    //   lanes 0-15: K = c0+{0..7,16..23}; lanes 16-31: K = c0+{8..15,24..31}
    const __bf16* aptr = Zb + (size_t)(m0 + l16) * CDIM + 8 * half;
    FragBF a[16];
    #pragma unroll
    for (int c = 0; c < 16; ++c) {
        a[c].u[0] = *(const uint4*)(aptr + c * 32);
        a[c].u[1] = *(const uint4*)(aptr + c * 32 + 16);
    }

    float bestv[8];
    int   bestk[8];
    #pragma unroll
    for (int j = 0; j < 8; j++) { bestv[j] = 3.0e38f; bestk[j] = 0; }

    for (int i = 0; i < 16; ++i) {
#if HAVE_TDM
        if (wave == 0) {
            if (i < 15) {   // issue next chunk, then guarantee current landed
                tdm_load_tile((unsigned)((i + 1) & 1) * B_BYTES,
                              Eb + (size_t)(i + 1) * 64 * CDIM, 64, KNUM);
                __builtin_amdgcn_s_wait_tensorcnt(1);
            } else {
                __builtin_amdgcn_s_wait_tensorcnt(0);
            }
        }
#endif
        __syncthreads();   // publish chunk i

        const char* bbuf = lds_raw + (unsigned)(i & 1) * B_BYTES;
        const int kc = i * 64;
        v8f acc[4] = {};
        float en[4];
        const char* brow[4];
        #pragma unroll
        for (int g = 0; g < 4; g++) {
            en[g]   = enorm[kc + g * 16 + l16];
            // ISA 16-bit B 32x16 layout: lanes0-15 c0+{0..7,16..23}; lanes16-31 c0+{8..15,24..31}
            brow[g] = bbuf + (unsigned)(g * 16 + l16) * ROWB + 16u * half;
        }

        #pragma unroll
        for (int c = 0; c < 16; ++c) {
            const int off = c * 64;          // 32 bf16 per c-step
            FragBF bb[4];
            #pragma unroll
            for (int g = 0; g < 4; g++) {    // batch all 8 LDS loads first
                bb[g].u[0] = *(const uint4*)(brow[g] + off);
                bb[g].u[1] = *(const uint4*)(brow[g] + off + 32);
            }
            #pragma unroll
            for (int g = 0; g < 4; g++) {    // then 4 back-to-back WMMAs
                acc[g] = __builtin_amdgcn_wmma_f32_16x16x32_bf16(
                    false, a[c].v, false, bb[g].v, (short)0, acc[g], false, false);
            }
        }

        // dist(n,k) ~ ||e_k||^2 - 2*S[n,k]  (||z||^2 is row-constant)
        #pragma unroll
        for (int g = 0; g < 4; g++) {
            int kk = kc + g * 16 + l16;
            #pragma unroll
            for (int j = 0; j < 8; j++) {
                float d = en[g] - 2.0f * acc[g][j];
                if (d < bestv[j]) { bestv[j] = d; bestk[j] = kk; }
            }
        }

        __syncthreads();   // all reads of buf[(i+1)&1] from pass i-1 are done
#if !HAVE_TDM
        if (i < 15) {
            stage_tile((unsigned)((i + 1) & 1) * B_BYTES,
                       Eb + (size_t)(i + 1) * 64 * CDIM, 64, threadIdx.x);
        }
#endif
    }

    // Cross-lane min-with-index inside each 16-lane half (xor 1,2,4,8).
    #pragma unroll
    for (int j = 0; j < 8; j++) {
        float v = bestv[j];
        int   k = bestk[j];
        #pragma unroll
        for (int off = 1; off < 16; off <<= 1) {
            float ov = __shfl_xor(v, off, 32);
            int   ok = __shfl_xor(k, off, 32);
            if (ov < v || (ov == v && ok < k)) { v = ov; k = ok; }
        }
        if (l16 == 0) idx_out[m0 + j + 8 * half] = k;
    }
}

// ---------------------------------------------------------------------------
// Kernel 4: quantized = E[idx] back to NCHW (coalesced writes) + idx as float
// ---------------------------------------------------------------------------
__global__ __launch_bounds__(256) void vq_gather(const float* __restrict__ E,
                                                 const int* __restrict__ idx,
                                                 float* __restrict__ out,
                                                 float* __restrict__ idxf) {
    int bc = blockIdx.x;             // b*512 + c
    int b = bc >> 9, c = bc & 511;
    int nbase = b << 10;
    float* o = out + (size_t)bc * 1024;
    for (int p = threadIdx.x; p < 1024; p += 256) {
        int k = idx[nbase + p];
        o[p] = E[(size_t)k * CDIM + c];
        if (c == 0) idxf[nbase + p] = (float)k;
    }
}

// ---------------------------------------------------------------------------
extern "C" void kernel_launch(void* const* d_in, const int* in_sizes, int n_in,
                              void* d_out, int out_size, void* d_ws, size_t ws_size,
                              hipStream_t stream) {
    const float* z = (const float*)d_in[0];   // [32,512,32,32] f32
    const float* E = (const float*)d_in[1];   // [1024,512] f32
    float* out = (float*)d_out;               // 16777216 quantized + 32768 idx

    char* ws = (char*)d_ws;
    size_t off = 0;
    __bf16* Eb    = (__bf16*)(ws + off); off += (size_t)KNUM * CDIM * 2;  // 1 MB
    float*  enorm = (float*)(ws + off);  off += (size_t)KNUM * 4;         // 4 KB
    int*    idx   = (int*)(ws + off);    off += (size_t)NPIX * 4;         // 128 KB
    __bf16* Zb    = (__bf16*)(ws + off);                                  // 33.5 MB

    vq_prep_embed<<<KNUM, 128, 0, stream>>>(E, Eb, enorm);
    vq_zconv<<<dim3(32, 16, 32), 256, 0, stream>>>(z, Zb);
    vq_main<<<NPIX / 128, 256, LDS_TOTAL, stream>>>(Zb, Eb, enorm, idx);
    vq_gather<<<32 * CDIM, 256, 0, stream>>>(E, idx, out, out + QELEMS);
}
